// LSTM_56092272886063
// MI455X (gfx1250) — matrix operand order repaired and checked by reading, not compiled, not run on hardware
//
#include <hip/hip_runtime.h>
#include <hip/hip_bf16.h>

typedef __bf16        v16bf __attribute__((ext_vector_type(16)));
typedef float         v8f   __attribute__((ext_vector_type(8)));
typedef unsigned int  v4u   __attribute__((ext_vector_type(4)));

union Frag { v16bf v; v4u q[2]; };

#define HID     128
#define NG      512     // 4*HID gates
#define BATCH   256
#define BM      16      // batch rows per workgroup
#define NWG     (BATCH / BM)
#define THREADS 256     // 8 wave32 waves

// Branchless activations: single v_exp_f32 + v_rcp_f32, no EXEC-modifying
// branches near the WMMA block, correct saturation at +/-inf.
__device__ __forceinline__ float sig_fast(float x) {
  return __fdividef(1.0f, 1.0f + __expf(-x));           // x->-inf: 1/inf = 0; x->+inf: 1
}
__device__ __forceinline__ float tanh_fast(float x) {
  float e = __expf(2.0f * x);                            // 0 .. inf
  return 1.0f - __fdividef(2.0f, e + 1.0f);              // -1 .. +1, branchless
}

// A fragment (16x32 bf16, MxK) from row-major LDS [BM][ktot].
// ISA 7.12.2: lane<16 -> row m=lane, K={k0..k0+7, k0+16..k0+23};
//             lane>=16 -> row m=lane-16, K={k0+8.., k0+24..}
__device__ __forceinline__ Frag load_afrag(const __hip_bfloat16* xh, int ktot, int k0, int lane) {
  Frag f;
  int m    = lane & 15;
  int koff = k0 + ((lane < 16) ? 0 : 8);
  f.q[0] = *(const v4u*)(xh + m * ktot + koff);
  f.q[1] = *(const v4u*)(xh + m * ktot + koff + 16);
  return f;
}

// B fragment (32x16 bf16, KxN). B column n == row n of wcomb (row-major [NG][ktot]).
// lane<16 -> col n, K = {k0..k0+7, k0+16..k0+23}; lane>=16 -> K = {k0+8.., k0+24..}
__device__ __forceinline__ Frag load_bfrag(const __hip_bfloat16* wc, int ktot, int n, int k0, int lane) {
  Frag f;
  int koff = k0 + ((lane < 16) ? 0 : 8);
  const __hip_bfloat16* p = wc + n * ktot + koff;
  f.q[0] = *(const v4u*)(p);
  f.q[1] = *(const v4u*)(p + 16);
  return f;
}

template<int KCH_IN, bool SRC_F32>
__launch_bounds__(THREADS, 1)
__global__ void lstm_layer_kernel(const float* __restrict__ xsrc_f,
                                  const __hip_bfloat16* __restrict__ xsrc_b,
                                  const float* __restrict__ w_ih,
                                  const float* __restrict__ w_hh,
                                  const float* __restrict__ b_ih,
                                  const float* __restrict__ b_hh,
                                  __hip_bfloat16* __restrict__ hbuf,
                                  int T, int kin) {
  constexpr int KIN_PAD = KCH_IN * 32;
  constexpr int KTOT    = KIN_PAD + HID;

  extern __shared__ char smem[];
  __hip_bfloat16* wcomb = (__hip_bfloat16*)smem;        // [NG][KTOT] bf16: [W_ih | W_hh]
  __hip_bfloat16* xh    = wcomb + NG * KTOT;            // [BM][KTOT] bf16: [x_t | h_{t-1}]
  float*          biasL = (float*)(xh + BM * KTOT);     // [NG]

  const int tid  = threadIdx.x;
  const int wv   = tid >> 5;       // wave id 0..7
  const int lane = tid & 31;
  const int r0   = blockIdx.x * BM;

  // ---- zero wcomb (for K padding) + xh (h_{-1} = 0) ----
  {
    unsigned int* z = (unsigned int*)smem;
    const int nz = (NG + BM) * KTOT / 2;
    for (int i = tid; i < nz; i += THREADS) z[i] = 0u;
  }
  __syncthreads();

  // ---- load weights fp32 -> bf16 LDS, combined [W_ih | W_hh] per gate-row ----
  for (int i = tid; i < NG * kin; i += THREADS) {
    int n = i / kin, c = i - n * kin;
    wcomb[n * KTOT + c] = __float2bfloat16(w_ih[i]);
  }
  for (int i = tid; i < NG * HID; i += THREADS) {
    int n = i >> 7, c = i & (HID - 1);
    wcomb[n * KTOT + KIN_PAD + c] = __float2bfloat16(w_hh[i]);
  }
  for (int i = tid; i < NG; i += THREADS) biasL[i] = b_ih[i] + b_hh[i];
  __syncthreads();

  // wave wv owns N-tiles {wv, 8+wv, 16+wv, 24+wv} => acc[0..3] = gates i,f,g,o
  // for hidden column j, same (m,n) lane positions across the 4 accumulators.
  const int j  = (lane & 15) + 16 * wv;   // hidden index 0..127
  const int nb = 16 * wv + (lane & 15);   // base gate row for this lane (q adds 128q)
  float bq[4];
  #pragma unroll
  for (int q = 0; q < 4; ++q) bq[q] = biasL[q * HID + nb];

  // recurrent B fragments resident in registers for the whole time loop
  Frag bR[4][4];
  #pragma unroll
  for (int q = 0; q < 4; ++q) {
    #pragma unroll
    for (int kc = 0; kc < 4; ++kc)
      bR[q][kc] = load_bfrag(wcomb, KTOT, q * HID + nb, KIN_PAD + kc * 32, lane);
  }

  // ---- stage x_0 into xh ----
  if constexpr (SRC_F32) {
    if (tid < BM * 8) {
      int i = tid >> 3, k = tid & 7;
      xh[i * KTOT + k] = __float2bfloat16(xsrc_f[((size_t)(r0 + i) * T) * 8 + k]);
    }
  } else {
    int i = tid >> 4, kb = (tid & 15) * 8;
    *(v4u*)(xh + i * KTOT + kb) = *(const v4u*)(xsrc_b + ((size_t)(r0 + i) * T) * HID + kb);
  }

  float cs[8];
  #pragma unroll
  for (int e = 0; e < 8; ++e) cs[e] = 0.0f;
  const int mbase = (lane < 16) ? 0 : 8;   // C/D layout: m = vgpr + 8*(lane>=16)
  __syncthreads();

  for (int t = 0; t < T; ++t) {
    // Phase 1: pull all A fragments [x_t | h_{t-1}] from LDS into registers
    Frag af[KCH_IN + 4];
    #pragma unroll
    for (int kc = 0; kc < KCH_IN + 4; ++kc)
      af[kc] = load_afrag(xh, KTOT, kc * 32, lane);
    __syncthreads();   // all xh reads retired before anyone overwrites it

    // Phase 2: z = [x_t|h] @ Wc^T (bf16 WMMA, fp32 accum; C starts at inline 0,
    // bias folded into the gate phase to avoid splat moves / WAR hazard NOPs)
    v8f acc[4];
    #pragma unroll
    for (int q = 0; q < 4; ++q) acc[q] = (v8f){0.f,0.f,0.f,0.f,0.f,0.f,0.f,0.f};
    #pragma unroll
    for (int kc = 0; kc < KCH_IN; ++kc) {
      #pragma unroll
      for (int q = 0; q < 4; ++q) {
        Frag b = load_bfrag(wcomb, KTOT, q * HID + nb, kc * 32, lane);
        acc[q] = __builtin_amdgcn_wmma_f32_16x16x32_bf16(
                   false, af[kc].v, false, b.v, (short)0, acc[q], false, false);
      }
    }
    #pragma unroll
    for (int kc = 0; kc < 4; ++kc) {
      #pragma unroll
      for (int q = 0; q < 4; ++q)
        acc[q] = __builtin_amdgcn_wmma_f32_16x16x32_bf16(
                   false, af[KCH_IN + kc].v, false, bR[q][kc].v, (short)0, acc[q], false, false);
    }

    // gates + state update (fully branchless); write h_t to LDS and global hbuf
    #pragma unroll
    for (int e = 0; e < 8; ++e) {
      float gi = sig_fast (acc[0][e] + bq[0]);
      float gf = sig_fast (acc[1][e] + bq[1]);
      float gg = tanh_fast(acc[2][e] + bq[2]);
      float go = sig_fast (acc[3][e] + bq[3]);
      cs[e] = gf * cs[e] + gi * gg;
      float hv = go * tanh_fast(cs[e]);
      int m = mbase + e;
      __hip_bfloat16 hb = __float2bfloat16(hv);
      xh[m * KTOT + KIN_PAD + j] = hb;
      hbuf[((size_t)(r0 + m) * T + t) * HID + j] = hb;
    }

    // stage x_{t+1} (in-place safe: slot t+1 is read here, written at step t+1)
    if (t + 1 < T) {
      if constexpr (SRC_F32) {
        if (tid < BM * 8) {
          int i = tid >> 3, k = tid & 7;
          xh[i * KTOT + k] = __float2bfloat16(xsrc_f[((size_t)(r0 + i) * T + (t + 1)) * 8 + k]);
        }
      } else {
        int i = tid >> 4, kb = (tid & 15) * 8;
        *(v4u*)(xh + i * KTOT + kb) =
            *(const v4u*)(xsrc_b + ((size_t)(r0 + i) * T + (t + 1)) * HID + kb);
      }
    }
    __syncthreads();   // h_t and x_{t+1} visible for next iteration's phase 1
  }
}

// y = leaky(h_last @ w1^T + b1) @ w2^T + b2   (tiny: 256 rows, one thread each)
__global__ void head_kernel(const __hip_bfloat16* __restrict__ hbuf,
                            const float* __restrict__ w1, const float* __restrict__ b1,
                            const float* __restrict__ w2, const float* __restrict__ b2,
                            float* __restrict__ out, int T) {
  int b = blockIdx.x * blockDim.x + threadIdx.x;
  if (b >= BATCH) return;
  const __hip_bfloat16* hp = hbuf + ((size_t)b * T + (T - 1)) * HID;
  float hv[HID];
  #pragma unroll 4
  for (int k = 0; k < HID; ++k) hv[k] = __bfloat162float(hp[k]);
  float mid[16];
  for (int u = 0; u < 16; ++u) {
    float s = b1[u];
    for (int k = 0; k < HID; ++k) s += hv[k] * w1[u * HID + k];
    mid[u] = (s >= 0.0f) ? s : 0.01f * s;
  }
  for (int o = 0; o < 4; ++o) {
    float s = b2[o];
    for (int u = 0; u < 16; ++u) s += mid[u] * w2[o * 16 + u];
    out[b * 4 + o] = s;
  }
}

static inline size_t lds_bytes(int ktot) {
  return (size_t)(NG + BM) * ktot * 2 + (size_t)NG * 4;
}

extern "C" void kernel_launch(void* const* d_in, const int* in_sizes, int n_in,
                              void* d_out, int out_size, void* d_ws, size_t ws_size,
                              hipStream_t stream) {
  const float* x      = (const float*)d_in[0];
  const float* wih[3] = {(const float*)d_in[1], (const float*)d_in[5], (const float*)d_in[9]};
  const float* whh[3] = {(const float*)d_in[2], (const float*)d_in[6], (const float*)d_in[10]};
  const float* bih[3] = {(const float*)d_in[3], (const float*)d_in[7], (const float*)d_in[11]};
  const float* bhh[3] = {(const float*)d_in[4], (const float*)d_in[8], (const float*)d_in[12]};
  const float* w1 = (const float*)d_in[13];
  const float* b1 = (const float*)d_in[14];
  const float* w2 = (const float*)d_in[15];
  const float* b2 = (const float*)d_in[16];

  const int T = in_sizes[0] / (BATCH * 8);
  __hip_bfloat16* hbuf = (__hip_bfloat16*)d_ws;   // [BATCH][T][HID] bf16 (64 MB)

  const size_t sh0 = lds_bytes(32 + HID);    // ~171 KB (layer 0, Kin=8 padded to 32)
  const size_t sh1 = lds_bytes(128 + HID);   // ~272 KB (layers 1-2) -- CDNA5 WGP LDS
  hipFuncSetAttribute(reinterpret_cast<const void*>(lstm_layer_kernel<1, true>),
                      hipFuncAttributeMaxDynamicSharedMemorySize, (int)sh0);
  hipFuncSetAttribute(reinterpret_cast<const void*>(lstm_layer_kernel<4, false>),
                      hipFuncAttributeMaxDynamicSharedMemorySize, (int)sh1);

  dim3 grid(NWG), blk(THREADS);
  lstm_layer_kernel<1, true ><<<grid, blk, sh0, stream>>>(x,       nullptr, wih[0], whh[0], bih[0], bhh[0], hbuf, T, 8);
  lstm_layer_kernel<4, false><<<grid, blk, sh1, stream>>>(nullptr, hbuf,    wih[1], whh[1], bih[1], bhh[1], hbuf, T, HID);
  lstm_layer_kernel<4, false><<<grid, blk, sh1, stream>>>(nullptr, hbuf,    wih[2], whh[2], bih[2], bhh[2], hbuf, T, HID);
  head_kernel<<<dim3(1), dim3(256), 0, stream>>>(hbuf, w1, b1, w2, b2, (float*)d_out, T);

  (void)n_in; (void)out_size; (void)ws_size;
}
